// GCNLayer_53154515255658
// MI455X (gfx1250) — compile-verified, matrix operand order checked by reading, not run on hardware
//
#include <hip/hip_runtime.h>
#include <stdint.h>

// ---------------------------------------------------------------------------
// GCN layer for MI455X (gfx1250, wave32, WMMA).
// N=8192, D_IN=128, D_HID=256, D_OUT=64
// Column range of the two N x N passes is split into JSEG segments for
// occupancy; partials are reduced in fixed order (deterministic, no atomics).
// ---------------------------------------------------------------------------

typedef __attribute__((ext_vector_type(16))) _Float16 v16h;
typedef __attribute__((ext_vector_type(8)))  _Float16 v8h;
typedef __attribute__((ext_vector_type(8)))  float    v8f;

constexpr int N     = 8192;
constexpr int DIN   = 128;
constexpr int DHID  = 256;
constexpr int DOUT  = 64;
constexpr int JSEG  = 8;          // column segments for the N x N passes
constexpr int JLEN  = N / JSEG;   // 1024 columns per segment

// Workspace layout (bytes, 4K-aligned blocks). Total ~31.4 MB.
constexpr size_t OFF_SCALE  = 0;         // 128 f32
constexpr size_t OFF_SHIFT  = 1024;      // 128 f32
constexpr size_t OFF_DVEC   = 4096;      // 8192 f32
constexpr size_t OFF_RSP    = 36864;     // JSEG x 8192 f32 partial row sums
constexpr size_t OFF_HN     = 299008;    // 8192x128 f16 (2 MB)
constexpr size_t OFF_WTT    = 2396160;   // 256x128 f16 (W_theta^T)
constexpr size_t OFF_WOT    = 2461696;   // 64x128 f16  (W_out^T)
constexpr size_t OFF_HX     = 2478080;   // 8192x256 f16 (4 MB)
constexpr size_t OFF_PT     = 6672384;   // 64x8192 f16 (P^T, 1 MB)
constexpr size_t OFF_MB     = 7720960;   // 8192*8192/32 u32 (8 MB)
constexpr size_t OFF_OUTP   = 16109568;  // JSEG x 8192 x 64 f32 partial out (16 MB)

// ---- WMMA helpers ----------------------------------------------------------

__device__ __forceinline__ v8f wmma16(v16h a, v16h b, v8f c) {
  return __builtin_amdgcn_wmma_f32_16x16x32_f16(
      /*neg_a=*/false, a, /*neg_b=*/false, b,
      /*c_mod=*/(short)0, c, /*reuse_a=*/false, /*reuse_b=*/false);
}

// Load a 16x32 f16 A-fragment (or the mirrored 32x16 B-fragment) from a
// row-major f16 matrix: lane&15 selects the row (M for A / N for B^T-storage),
// per-lane K chunks live at kb+(lane>>4)*8 and kb+16+(lane>>4)*8.
__device__ __forceinline__ v16h load_frag(const _Float16* __restrict__ base,
                                          int row0, int ld, int kb) {
  int lane = threadIdx.x & 31;
  int r  = row0 + (lane & 15);
  int hb = (lane >> 4) * 8;
  const _Float16* p = base + (size_t)r * ld + kb + hb;
  v8h lo = *(const v8h*)(p);
  v8h hi = *(const v8h*)(p + 16);
  return __builtin_shufflevector(lo, hi, 0,1,2,3,4,5,6,7,8,9,10,11,12,13,14,15);
}

// ---- Stage 1: batch-norm statistics ---------------------------------------

__global__ __launch_bounds__(256) void k_bnstats(
    const float* __restrict__ H, const float* __restrict__ bnw,
    const float* __restrict__ bnb, float* __restrict__ scale,
    float* __restrict__ shift) {
  __shared__ float s1[256], s2[256];
  int d = blockIdx.x, t = threadIdx.x;
  float a = 0.f, b = 0.f;
  for (int r = t; r < N; r += 256) {
    float x = H[(size_t)r * DIN + d];
    a += x; b += x * x;
  }
  s1[t] = a; s2[t] = b;
  __syncthreads();
  for (int off = 128; off > 0; off >>= 1) {
    if (t < off) { s1[t] += s1[t + off]; s2[t] += s2[t + off]; }
    __syncthreads();
  }
  if (t == 0) {
    float mean = s1[0] / (float)N;
    float var  = s2[0] / (float)N - mean * mean;
    float sc   = bnw[d] * rsqrtf(var + 1e-5f);
    scale[d] = sc;
    shift[d] = bnb[d] - mean * sc;
  }
}

// ---- Stage 2: normalized H in fp16 ----------------------------------------

__global__ __launch_bounds__(256) void k_hn(
    const float* __restrict__ H, const float* __restrict__ scale,
    const float* __restrict__ shift, _Float16* __restrict__ Hn) {
  int idx = blockIdx.x * 256 + threadIdx.x;
  if (idx < N * DIN) {
    int d = idx & (DIN - 1);
    Hn[idx] = (_Float16)(H[idx] * scale[d] + shift[d]);
  }
}

// ---- Stage 2b: transpose+convert weights to fp16 --------------------------

__global__ __launch_bounds__(256) void k_wcvt(
    const float* __restrict__ Wt, const float* __restrict__ Wo,
    _Float16* __restrict__ Wtt, _Float16* __restrict__ Wot) {
  int idx = blockIdx.x * 256 + threadIdx.x;
  if (idx < DIN * DHID) {                 // W_theta^T [256,128]
    int n = idx >> 7, k = idx & 127;
    Wtt[idx] = (_Float16)Wt[(size_t)k * DHID + n];
  } else if (idx < DIN * DHID + DIN * DOUT) {  // W_out^T [64,128]
    int i2 = idx - DIN * DHID;
    int n = i2 >> 7, k = i2 & 127;
    Wot[i2] = (_Float16)Wo[(size_t)k * DOUT + n];
  }
}

// ---- Stage 3: adjacency mask -> bitmask (one read of the 268MB A) ---------

__global__ __launch_bounds__(256) void k_mask(
    const float* __restrict__ A, uint32_t* __restrict__ mb) {
  int w = blockIdx.x * 256 + threadIdx.x;   // one 32-bit word per thread
  const float4* A4 = (const float4*)(A + (size_t)w * 32);
  uint32_t bits = 0;
#pragma unroll
  for (int q = 0; q < 8; ++q) {
    float4 f = A4[q];
    int b = q * 4;
    if (ceilf(f.x * 1e-5f) > 0.f) bits |= 1u << b;
    if (ceilf(f.y * 1e-5f) > 0.f) bits |= 1u << (b + 1);
    if (ceilf(f.z * 1e-5f) > 0.f) bits |= 1u << (b + 2);
    if (ceilf(f.w * 1e-5f) > 0.f) bits |= 1u << (b + 3);
  }
  mb[w] = bits;
}

// ---- Stage 4: Hx = Hn@W_theta + b_theta (fp16), P^T = (Hn@W_out + b_out)^T -

__global__ __launch_bounds__(256) void k_inproj(
    const _Float16* __restrict__ Hn, const _Float16* __restrict__ Wtt,
    const _Float16* __restrict__ Wot, const float* __restrict__ bt,
    const float* __restrict__ bo, _Float16* __restrict__ Hx,
    _Float16* __restrict__ Pt) {
  int wave = threadIdx.x >> 5;
  int lane = threadIdx.x & 31;
  int nl = lane & 15, hi = lane >> 4;
  int i0 = (blockIdx.x * 8 + wave) * 16;

  v16h a[4];
#pragma unroll
  for (int kc = 0; kc < 4; ++kc) a[kc] = load_frag(Hn, i0, DIN, kc * 32);

#pragma unroll 4
  for (int nt = 0; nt < DHID / 16; ++nt) {
    v8f c = {};
#pragma unroll
    for (int kc = 0; kc < 4; ++kc)
      c = wmma16(a[kc], load_frag(Wtt, nt * 16, DIN, kc * 32), c);
    int n = nt * 16 + nl;
    float bv = bt[n];
#pragma unroll
    for (int v = 0; v < 8; ++v)
      Hx[(size_t)(i0 + v + 8 * hi) * DHID + n] = (_Float16)(c[v] + bv);
  }
#pragma unroll
  for (int nt = 0; nt < DOUT / 16; ++nt) {
    v8f c = {};
#pragma unroll
    for (int kc = 0; kc < 4; ++kc)
      c = wmma16(a[kc], load_frag(Wot, nt * 16, DIN, kc * 32), c);
    int n = nt * 16 + nl;
    float bv = bo[n];
#pragma unroll
    for (int v = 0; v < 8; ++v)   // store transposed: Pt[n][row]
      Pt[(size_t)n * N + i0 + v + 8 * hi] = (_Float16)(c[v] + bv);
  }
}

// ---- Stage 5 (pass 1): partial row sums of Ad over one column segment -----
// grid = (N/128) * JSEG blocks; each wave owns a 16-row strip within one
// segment -> complete segment-partial row sums locally, no atomics.

__global__ __launch_bounds__(256) void k_rowsum(
    const _Float16* __restrict__ Hx, const uint32_t* __restrict__ mb,
    float* __restrict__ rsp) {
  int jseg = blockIdx.x & (JSEG - 1);
  int bs   = blockIdx.x / JSEG;
  int wave = threadIdx.x >> 5;
  int lane = threadIdx.x & 31;
  int nl = lane & 15, hi = lane >> 4;
  int i0 = (bs * 8 + wave) * 16;
  int jlo = jseg * JLEN;

  v16h a[8];
#pragma unroll
  for (int kc = 0; kc < 8; ++kc) a[kc] = load_frag(Hx, i0, DHID, kc * 32);

  float rs[8] = {0.f, 0.f, 0.f, 0.f, 0.f, 0.f, 0.f, 0.f};
  for (int jt = jlo; jt < jlo + JLEN; jt += 16) {
    v8f c = {};
#pragma unroll
    for (int kc = 0; kc < 8; ++kc)
      c = wmma16(a[kc], load_frag(Hx, jt, DHID, kc * 32), c);
    int col = jt + nl;
    int wi = col >> 5, sh = col & 31;
#pragma unroll
    for (int v = 0; v < 8; ++v) {
      int m = i0 + v + 8 * hi;
      float s = fmaxf(1.f / (1.f + __expf(-c[v])), 0.1f);
      uint32_t mw = mb[m * (N / 32) + wi];
      float ad = ((mw >> sh) & 1u) ? s : 0.f;
      if (col == m) ad += 1.f;       // identity lands in its own segment
      rs[v] += ad;
    }
  }
  // reduce the 16 column-lanes of each half-wave
#pragma unroll
  for (int v = 0; v < 8; ++v) {
    rs[v] += __shfl_xor(rs[v], 1, 32);
    rs[v] += __shfl_xor(rs[v], 2, 32);
    rs[v] += __shfl_xor(rs[v], 4, 32);
    rs[v] += __shfl_xor(rs[v], 8, 32);
  }
  if (nl == 0) {
#pragma unroll
    for (int v = 0; v < 8; ++v)
      rsp[(size_t)jseg * N + i0 + v + 8 * hi] = rs[v];
  }
}

// ---- Stage 5b: d = rsqrt(sum of segment partials), fixed order ------------

__global__ __launch_bounds__(256) void k_dvec(
    const float* __restrict__ rsp, float* __restrict__ dvec) {
  int i = blockIdx.x * 256 + threadIdx.x;
  if (i < N) {
    float s = 0.f;
#pragma unroll
    for (int q = 0; q < JSEG; ++q) s += rsp[(size_t)q * N + i];
    dvec[i] = rsqrtf(s);
  }
}

// ---- Stage 6 (pass 2): A_hat tiles + fused partial out = A_hat @ P --------

__global__ __launch_bounds__(256) void k_ahat_out(
    const _Float16* __restrict__ Hx, const _Float16* __restrict__ Pt,
    const uint32_t* __restrict__ mb, const float* __restrict__ dvec,
    float* __restrict__ outp, float* __restrict__ Ahat) {
  __shared__ _Float16 lt[8][16 * 32];   // per-wave 16x32 fp16 A_hat tile
  int jseg = blockIdx.x & (JSEG - 1);
  int bs   = blockIdx.x / JSEG;
  int wave = threadIdx.x >> 5;
  int lane = threadIdx.x & 31;
  int nl = lane & 15, hi = lane >> 4;
  int i0 = (bs * 8 + wave) * 16;
  int jlo = jseg * JLEN;
  _Float16* myt = lt[wave];

  v16h a[8];
#pragma unroll
  for (int kc = 0; kc < 8; ++kc) a[kc] = load_frag(Hx, i0, DHID, kc * 32);
  float di[8];
#pragma unroll
  for (int v = 0; v < 8; ++v) di[v] = dvec[i0 + v + 8 * hi];

  v8f zero = {};
  v8f oa[4] = {zero, zero, zero, zero};

  for (int j0 = jlo; j0 < jlo + JLEN; j0 += 32) {
    v8f c0 = {}, c1 = {};
#pragma unroll
    for (int kc = 0; kc < 8; ++kc) {
      c0 = wmma16(a[kc], load_frag(Hx, j0,      DHID, kc * 32), c0);
      c1 = wmma16(a[kc], load_frag(Hx, j0 + 16, DHID, kc * 32), c1);
    }
    float dj0 = dvec[j0 + nl];
    float dj1 = dvec[j0 + 16 + nl];
#pragma unroll
    for (int v = 0; v < 8; ++v) {
      int m = i0 + v + 8 * hi;
      uint32_t mw = mb[m * (N / 32) + (j0 >> 5)];  // covers both 16-col tiles
      float s0 = fmaxf(1.f / (1.f + __expf(-c0[v])), 0.1f);
      float s1 = fmaxf(1.f / (1.f + __expf(-c1[v])), 0.1f);
      float ad0 = ((mw >> nl) & 1u) ? s0 : 0.f;
      float ad1 = ((mw >> (16 + nl)) & 1u) ? s1 : 0.f;
      if (j0 + nl == m)      ad0 += 1.f;
      if (j0 + 16 + nl == m) ad1 += 1.f;
      float ah0 = ad0 * di[v] * dj0;
      float ah1 = ad1 * di[v] * dj1;
      size_t ro = (size_t)m * N + j0;
      Ahat[ro + nl]      = ah0;
      Ahat[ro + 16 + nl] = ah1;
      int lr = (v + 8 * hi) * 32;           // repack C/D layout -> row-major
      myt[lr + nl]      = (_Float16)ah0;
      myt[lr + 16 + nl] = (_Float16)ah1;
    }
    // Rebuild as A-fragment (same-wave LDS, in-order: no barrier needed)
    v16h af;
    int base = nl * 32 + hi * 8;
#pragma unroll
    for (int i = 0; i < 8; ++i) {
      af[i]     = myt[base + i];
      af[i + 8] = myt[base + 16 + i];
    }
#pragma unroll
    for (int t = 0; t < 4; ++t)
      oa[t] = wmma16(af, load_frag(Pt, t * 16, N, j0), oa[t]);
  }
  // store segment-partial out (no activation yet)
  float* op = outp + (size_t)jseg * N * DOUT;
#pragma unroll
  for (int t = 0; t < 4; ++t) {
#pragma unroll
    for (int v = 0; v < 8; ++v)
      op[(size_t)(i0 + v + 8 * hi) * DOUT + t * 16 + nl] = oa[t][v];
  }
}

// ---- Stage 6b: reduce out partials (fixed order) + LeakyReLU --------------

__global__ __launch_bounds__(256) void k_outred(
    const float* __restrict__ outp, float* __restrict__ out) {
  int idx = blockIdx.x * 256 + threadIdx.x;
  if (idx < N * DOUT) {
    float o = 0.f;
#pragma unroll
    for (int q = 0; q < JSEG; ++q) o += outp[(size_t)q * N * DOUT + idx];
    out[idx] = (o >= 0.f) ? o : 0.01f * o;
  }
}

// ---------------------------------------------------------------------------

extern "C" void kernel_launch(void* const* d_in, const int* in_sizes, int n_in,
                              void* d_out, int out_size, void* d_ws, size_t ws_size,
                              hipStream_t stream) {
  const float* H   = (const float*)d_in[0];
  const float* A   = (const float*)d_in[1];
  const float* bnw = (const float*)d_in[2];
  const float* bnb = (const float*)d_in[3];
  const float* Wt  = (const float*)d_in[4];
  const float* bt  = (const float*)d_in[5];
  const float* Wo  = (const float*)d_in[6];
  const float* bo  = (const float*)d_in[7];

  float* out  = (float*)d_out;
  float* Ahat = out + (size_t)N * DOUT;

  char* ws = (char*)d_ws;
  float*     scale  = (float*)(ws + OFF_SCALE);
  float*     shift  = (float*)(ws + OFF_SHIFT);
  float*     dvec   = (float*)(ws + OFF_DVEC);
  float*     rsp    = (float*)(ws + OFF_RSP);
  _Float16*  Hn     = (_Float16*)(ws + OFF_HN);
  _Float16*  Wtt    = (_Float16*)(ws + OFF_WTT);
  _Float16*  Wot    = (_Float16*)(ws + OFF_WOT);
  _Float16*  Hx     = (_Float16*)(ws + OFF_HX);
  _Float16*  Pt     = (_Float16*)(ws + OFF_PT);
  uint32_t*  mb     = (uint32_t*)(ws + OFF_MB);
  float*     outp   = (float*)(ws + OFF_OUTP);

  k_bnstats<<<DIN, 256, 0, stream>>>(H, bnw, bnb, scale, shift);
  k_hn<<<(N * DIN) / 256, 256, 0, stream>>>(H, scale, shift, Hn);
  k_wcvt<<<(DIN * DHID + DIN * DOUT + 255) / 256, 256, 0, stream>>>(Wt, Wo, Wtt, Wot);
  k_mask<<<(N / 32) * (N / 256), 256, 0, stream>>>(A, mb);         // 8192 blocks
  k_inproj<<<N / 128, 256, 0, stream>>>(Hn, Wtt, Wot, bt, bo, Hx, Pt);
  k_rowsum<<<(N / 128) * JSEG, 256, 0, stream>>>(Hx, mb, rsp);     // 512 blocks
  k_dvec<<<N / 256, 256, 0, stream>>>(rsp, dvec);
  k_ahat_out<<<(N / 128) * JSEG, 256, 0, stream>>>(Hx, Pt, mb, dvec, outp, Ahat);
  k_outred<<<(N * DOUT) / 256, 256, 0, stream>>>(outp, out);
}
